// CascadedAttention_9964324127542
// MI455X (gfx1250) — compile-verified, matrix operand order checked by reading, not run on hardware
//
#include <hip/hip_runtime.h>
#include <hip/hip_bf16.h>

#define NB 64            // B
#define NT 512           // T
#define ND 1024          // D
#define NV 28            // V
#define VP 32            // V padded to WMMA tile width
#define NM (NB * NT)     // 32768 GEMM rows

typedef __attribute__((ext_vector_type(16))) __bf16 v16bf;
typedef __attribute__((ext_vector_type(8)))  float  v8f;

// round-to-nearest-even f32 -> bf16, packed pair into one dword
static __device__ __forceinline__ unsigned pk_bf16(float a, float b) {
    unsigned ua = __float_as_uint(a);
    unsigned ub = __float_as_uint(b);
    ua += 0x7FFFu + ((ua >> 16) & 1u);
    ub += 0x7FFFu + ((ub >> 16) & 1u);
    return (ua >> 16) | (ub & 0xFFFF0000u);
}

// ---------------- csum[b,d] = sum_t x[b,t,d]  (context is step-invariant) ----
__global__ void colsum_k(const float* __restrict__ x, float* __restrict__ csum) {
    int i = blockIdx.x * blockDim.x + threadIdx.x;
    if (i >= NB * ND) return;
    int b = i / ND, d = i % ND;
    const float* p = x + (size_t)b * NT * ND + d;
    float s = 0.f;
    for (int t = 0; t < NT; ++t) s += p[(size_t)t * ND];
    csum[i] = s;
}

// ------- cCo[b,v] = csum[b]·Co[:,v];  ewo[r] = emb_table[r,:]·Wo[:,0] --------
__global__ void prep_k(const float* __restrict__ csum, const float* __restrict__ Co,
                       const float* __restrict__ emb,  const float* __restrict__ Wo,
                       float* __restrict__ cCo, float* __restrict__ ewo) {
    int b = blockIdx.x;
    int v = threadIdx.x;            // 0..31
    if (b < NB) {
        float s = 0.f;
        if (v < NV) {
            const float* cs = csum + (size_t)b * ND;
            for (int d = 0; d < ND; ++d) s += cs[d] * Co[(size_t)d * NV + v];
        }
        cCo[b * VP + v] = s;
    } else {
        if (v < NV) {
            float s = 0.f;
            for (int k = 0; k < NV; ++k) s += emb[v * NV + k] * Wo[k];
            ewo[v] = s;
        }
    }
}

// --------- G[m, 0:32] = x_flat[m, :] @ Uo_padded   (bf16 WMMA, f32 acc) ------
// Block: 256 thr = 8 waves; wave -> 16x32 output tile; block -> 128 rows.
__global__ __launch_bounds__(256)
void gemm_k(const float* __restrict__ x, const float* __restrict__ Uo,
            float* __restrict__ G) {
    // Uo columns pre-packed as bf16 pairs; pitch 513 dwords avoids 64-bank alias.
    __shared__ unsigned ldsB[NV * 513];   // 28*513*4 = 57456 B
    const int tid = threadIdx.x;

    for (int i = tid; i < NV * 512; i += 256) {
        int v = i >> 9;          // column 0..27
        int p = i & 511;         // k-pair index
        int k = p * 2;
        ldsB[v * 513 + p] = pk_bf16(Uo[(size_t)k * NV + v], Uo[(size_t)(k + 1) * NV + v]);
    }
    __syncthreads();

    const int wave = tid >> 5;
    const int lane = tid & 31;
    const int m    = lane & 15;          // M row (A) / N col (B) within tile
    const int kh   = lane >> 4;          // K half selector
    const int rowBase = blockIdx.x * 128 + wave * 16;

    const float* xr = x + (size_t)(rowBase + m) * ND;
    const unsigned* b0p = &ldsB[m * 513];
    const int c1 = 16 + m;
    const unsigned* b1p = &ldsB[((c1 < NV) ? c1 : 0) * 513];  // masked below

    v8f acc0 = {};
    v8f acc1 = {};

    for (int kb = 0; kb < ND; kb += 32) {
        __builtin_prefetch(xr + kb + 32, 0, 1);    // global_prefetch_b8
        union { v16bf v; unsigned u[8]; } A, B0, B1;
        #pragma unroll
        for (int j = 0; j < 8; ++j) {
            // 16-bit A layout: VGPR j holds K pair (j/4)*16 + kh*8 + (j%4)*2
            int koff = ((j >> 2) << 4) + (kh << 3) + ((j & 3) << 1);
            int k = kb + koff;
            float2 xa = *(const float2*)(xr + k);
            A.u[j] = pk_bf16(xa.x, xa.y);
            int p = k >> 1;
            B0.u[j] = b0p[p];
            unsigned bv = b1p[p];
            B1.u[j] = (c1 < NV) ? bv : 0u;         // zero-pad dead columns 28..31
        }
        acc0 = __builtin_amdgcn_wmma_f32_16x16x32_bf16(
                   false, A.v, false, B0.v, (short)0, acc0, false, false);
        acc1 = __builtin_amdgcn_wmma_f32_16x16x32_bf16(
                   false, A.v, false, B1.v, (short)0, acc1, false, false);
    }

    // C/D layout: VGPR r -> row r (lanes 0-15) or row 8+r (lanes 16-31), col = lane%16
    const int rext = kh * 8;
    float* gp = G + (size_t)rowBase * VP;
    #pragma unroll
    for (int r = 0; r < 8; ++r) {
        int row = r + rext;
        gp[(size_t)row * VP + m]      = acc0[r];
        gp[(size_t)row * VP + 16 + m] = acc1[r];
    }
}

// ---- per-(b,v) recurrence: y_t = sigmoid(ewo[(int)y_{t-1}] + G[t-1] + cCo) --
__global__ void scan_k(const float* __restrict__ G, const float* __restrict__ cCo,
                       const float* __restrict__ ewo, float* __restrict__ out) {
    int i = blockIdx.x * blockDim.x + threadIdx.x;
    if (i >= NB * NV) return;
    int b = i / NV, v = i % NV;
    const float bias = cCo[b * VP + v];
    const float* g = G + (size_t)b * NT * VP + v;
    float* o = out + (size_t)b * NT * NV + v;
    float y = 0.f;
    for (int t = 0; t < NT; ++t) {
        int idx = (int)y;
        idx = idx < 0 ? 0 : (idx > NV - 1 ? NV - 1 : idx);
        float hu = (t == 0) ? 0.f : g[(size_t)(t - 1) * VP];
        float z = ewo[idx] + hu + bias;
        y = 1.f / (1.f + __expf(-z));
        o[(size_t)t * NV] = y;
    }
}

extern "C" void kernel_launch(void* const* d_in, const int* in_sizes, int n_in,
                              void* d_out, int out_size, void* d_ws, size_t ws_size,
                              hipStream_t stream) {
    const float* x   = (const float*)d_in[0];
    // d_in[1]=Wa, d_in[2]=Ua, d_in[3]=Va: dead (softmax over size-1 axis -> ones)
    const float* Wo  = (const float*)d_in[4];
    const float* Uo  = (const float*)d_in[5];
    const float* Co  = (const float*)d_in[6];
    const float* emb = (const float*)d_in[7];
    float* out = (float*)d_out;

    char*  ws   = (char*)d_ws;
    float* G    = (float*)ws;                                  // NM*VP f32 = 4 MB
    float* csum = (float*)(ws + (size_t)NM * VP * sizeof(float)); // NB*ND
    float* cCo  = csum + (size_t)NB * ND;                      // NB*VP
    float* ewo  = cCo + (size_t)NB * VP;                       // VP

    colsum_k<<<(NB * ND + 255) / 256, 256, 0, stream>>>(x, csum);
    prep_k  <<<NB + 1, 32, 0, stream>>>(csum, Co, emb, Wo, cCo, ewo);
    gemm_k  <<<NM / 128, 256, 0, stream>>>(x, Uo, G);
    scan_k  <<<(NB * NV + 255) / 256, 256, 0, stream>>>(G, cCo, ewo, out);
}
